// CRF_79551384256819
// MI455X (gfx1250) — matrix operand order repaired and checked by reading, not compiled, not run on hardware
//
#include <hip/hip_runtime.h>
#include <hip/hip_bf16.h>
#include <math.h>

// CRF log-likelihood for (B,T,L) = (64,1024,256) on gfx1250.
// Denominator forward recursion recast as per-step GEMM via exp-rescaling,
// executed with v_wmma_f32_16x16x32_f16. E=exp(trans) lives in registers.

typedef __attribute__((ext_vector_type(16))) _Float16 v16h;
typedef __attribute__((ext_vector_type(8)))  _Float16 v8h;
typedef __attribute__((ext_vector_type(8)))  float    v8f;

static constexpr int kB = 64;
static constexpr int kT = 1024;
static constexpr int kL = 256;

static constexpr int SSTR = 260;  // score row stride (floats): 260 % 64 == 4 -> conflict-free rows
static constexpr int ASTR = 264;  // Abuf row stride (halves): 528 B, 16B-aligned rows, 4-bank skew

// ---------------------------------------------------------------------------
// Kernel 1: numerator (gather + reduce), one block per batch element.
// ---------------------------------------------------------------------------
__global__ __launch_bounds__(256)
void crf_num_kernel(const float* __restrict__ h, const int* __restrict__ labels,
                    const unsigned char* __restrict__ mask, const float* __restrict__ trans,
                    const float* __restrict__ startv, const float* __restrict__ endv,
                    float* __restrict__ num) {
  const int b   = blockIdx.x;
  const int tid = threadIdx.x;
  __shared__ float redf[256];
  __shared__ int   redi[256];

  const int*           lab = labels + (size_t)b * kT;
  const unsigned char* mk  = mask   + (size_t)b * kT;

  float acc = 0.0f;
  int   mcount = 0;
  for (int t = tid; t < kT; t += 256) {
    const int mt = mk[t] ? 1 : 0;
    mcount += mt;
    if (t < kT - 1) {
      const int   yt   = lab[t];
      const int   yt1  = lab[t + 1];
      const float emit = h[((size_t)b * kT + t) * kL + yt];
      const float tr   = trans[yt * kL + yt1];
      acc += emit * (float)mt + tr * (mk[t + 1] ? 1.0f : 0.0f);
    }
  }
  redf[tid] = acc;
  redi[tid] = mcount;
  __syncthreads();
  for (int s = 128; s > 0; s >>= 1) {
    if (tid < s) { redf[tid] += redf[tid + s]; redi[tid] += redi[tid + s]; }
    __syncthreads();
  }
  if (tid == 0) {
    const int   last_idx = redi[0] - 1;
    const int   ll       = lab[last_idx];
    const float mlast    = mk[kT - 1] ? 1.0f : 0.0f;
    num[b] = startv[lab[0]] + redf[0]
           + h[((size_t)b * kT + (kT - 1)) * kL + ll] * mlast + endv[ll];
  }
}

// ---------------------------------------------------------------------------
// Kernel 2: forward recursion. One block per 16-row batch tile (4 blocks).
// 256 threads = 8 wave32s; wave w owns output columns [32w, 32w+32).
// ---------------------------------------------------------------------------
__global__ __launch_bounds__(256, 1)
void crf_scan_kernel(const float* __restrict__ h, const unsigned char* __restrict__ mask,
                     const float* __restrict__ trans, const float* __restrict__ startv,
                     const float* __restrict__ endv, const float* __restrict__ num,
                     float* __restrict__ out) {
  __shared__ float score[16 * SSTR];
  __shared__ alignas(16) _Float16 Abuf[16 * ASTR];
  __shared__ float rowmax[16];

  const int b0   = blockIdx.x * 16;
  const int tid  = threadIdx.x;
  const int lane = tid & 31;
  const int wv   = tid >> 5;        // wave id 0..7
  const int hi   = lane >> 4;       // half-wave select
  const int nl   = lane & 15;       // lane within half
  const int n0   = wv * 32;         // first column of this wave's 2 N-tiles

  // phase-1 mapping: 16 lanes per row
  const int m1 = tid >> 4;
  const int s1 = tid & 15;

  // --- Preload B fragments: E[k][n] = exp(trans[k][n]), resident in VGPRs. ---
  // B, k order used here (same as A): lanes 0-15 K=0..15, lanes 16-31 K=16..31; VGPR v = K pair (2v,2v+1).
  v16h Breg[2][8];
#pragma unroll
  for (int tile = 0; tile < 2; ++tile) {
    const int n = n0 + tile * 16 + nl;
#pragma unroll
    for (int kc = 0; kc < 8; ++kc) {
      const int kb = kc * 32 + hi * 16;
      v16h bf;
#pragma unroll
      for (int v = 0; v < 8; ++v) {
        const int k0 = kb + 2 * v;
        bf[2 * v]     = (_Float16)__expf(trans[k0 * kL + n]);
        bf[2 * v + 1] = (_Float16)__expf(trans[(k0 + 1) * kL + n]);
      }
      Breg[tile][kc] = bf;
    }
  }

  // --- init: score[m][j] = start[j] + h[b0+m][0][j] ---
  for (int idx = tid; idx < 16 * kL; idx += 256) {
    const int m = idx >> 8;
    const int j = idx & 255;
    score[m * SSTR + j] = startv[j] + h[((size_t)(b0 + m) * kT) * kL + j];
  }
  __syncthreads();

  for (int t = 1; t < kT; ++t) {
    // Issue h (and mask) loads early; consumed after the WMMA chain.
    float hreg0[8], hreg1[8];
    unsigned char mreg[8];
#pragma unroll
    for (int r = 0; r < 8; ++r) {
      const int    m    = r + hi * 8;
      const size_t base = ((size_t)(b0 + m) * kT + t) * (size_t)kL;
      hreg0[r] = h[base + n0 + nl];
      hreg1[r] = h[base + n0 + 16 + nl];
      mreg[r]  = mask[(size_t)(b0 + m) * kT + t];
    }

    // --- phase 1: per-row max (shuffle reduce over 16 lanes), exp into Abuf f16 ---
    float vmax = score[m1 * SSTR + s1 * 16];
#pragma unroll
    for (int k = 1; k < 16; ++k) vmax = fmaxf(vmax, score[m1 * SSTR + s1 * 16 + k]);
#pragma unroll
    for (int off = 8; off > 0; off >>= 1) vmax = fmaxf(vmax, __shfl_xor(vmax, off, 16));
    if (s1 == 0) rowmax[m1] = vmax;
#pragma unroll
    for (int k = 0; k < 16; ++k) {
      const float p = __expf(score[m1 * SSTR + s1 * 16 + k] - vmax);
      Abuf[m1 * ASTR + s1 * 16 + k] = (_Float16)p;
    }
    __syncthreads();  // Abuf + rowmax ready; all score reads for step t done

    // --- phase 2: q = p @ E via 16 WMMAs, dual accumulators break C chain ---
    v8f c0a = {}, c0b = {}, c1a = {}, c1b = {};
#pragma unroll
    for (int kc = 0; kc < 8; ++kc) {
      const int kb = kc * 32;
      // A 16x32 f16 layout: lanes M=nl; K = hi*16..hi*16+15 (contiguous, matches B).
      const v8h alo = *(const v8h*)&Abuf[nl * ASTR + kb + hi * 16];
      const v8h ahi = *(const v8h*)&Abuf[nl * ASTR + kb + hi * 16 + 8];
      const v16h af = __builtin_shufflevector(alo, ahi, 0, 1, 2, 3, 4, 5, 6, 7,
                                              8, 9, 10, 11, 12, 13, 14, 15);
      if (kc & 1) {
        c0b = __builtin_amdgcn_wmma_f32_16x16x32_f16(false, af, false, Breg[0][kc],
                                                     (short)0, c0b, false, false);
        c1b = __builtin_amdgcn_wmma_f32_16x16x32_f16(false, af, false, Breg[1][kc],
                                                     (short)0, c1b, false, false);
      } else {
        c0a = __builtin_amdgcn_wmma_f32_16x16x32_f16(false, af, false, Breg[0][kc],
                                                     (short)0, c0a, false, false);
        c1a = __builtin_amdgcn_wmma_f32_16x16x32_f16(false, af, false, Breg[1][kc],
                                                     (short)0, c1a, false, false);
      }
    }
    const v8f c0 = c0a + c0b;
    const v8f c1 = c1a + c1b;

    // --- phase 3: score_new = rowmax + log(q) + h_t (masked update) ---
#pragma unroll
    for (int r = 0; r < 8; ++r) {
      const int   m  = r + hi * 8;   // C layout: VGPR r -> M = r (+8 for upper lanes)
      const float rm = rowmax[m];
      if (mreg[r]) {
        score[m * SSTR + n0 + nl]      = rm + __logf(c0[r]) + hreg0[r];
        score[m * SSTR + n0 + 16 + nl] = rm + __logf(c1[r]) + hreg1[r];
      }
    }
    __syncthreads();  // score ready for step t+1
  }

  // --- final: denom = logsumexp_j(score + end); out = num - denom ---
  {
    float vals[16];
#pragma unroll
    for (int k = 0; k < 16; ++k)
      vals[k] = score[m1 * SSTR + s1 * 16 + k] + endv[s1 * 16 + k];
    float vmax = vals[0];
#pragma unroll
    for (int k = 1; k < 16; ++k) vmax = fmaxf(vmax, vals[k]);
#pragma unroll
    for (int off = 8; off > 0; off >>= 1) vmax = fmaxf(vmax, __shfl_xor(vmax, off, 16));
    float se = 0.0f;
#pragma unroll
    for (int k = 0; k < 16; ++k) se += __expf(vals[k] - vmax);
#pragma unroll
    for (int off = 8; off > 0; off >>= 1) se += __shfl_xor(se, off, 16);
    if (s1 == 0) out[b0 + m1] = num[b0 + m1] - (vmax + __logf(se));
  }
}

// ---------------------------------------------------------------------------
extern "C" void kernel_launch(void* const* d_in, const int* in_sizes, int n_in,
                              void* d_out, int out_size, void* d_ws, size_t ws_size,
                              hipStream_t stream) {
  (void)in_sizes; (void)n_in; (void)out_size; (void)ws_size;
  const float*         h      = (const float*)d_in[0];
  const int*           labels = (const int*)d_in[1];
  const unsigned char* mask   = (const unsigned char*)d_in[2];
  const float*         trans  = (const float*)d_in[3];
  const float*         startv = (const float*)d_in[4];
  const float*         endv   = (const float*)d_in[5];
  float*               out    = (float*)d_out;
  float*               num    = (float*)d_ws;  // 64 floats of scratch

  crf_num_kernel<<<kB, 256, 0, stream>>>(h, labels, mask, trans, startv, endv, num);
  crf_scan_kernel<<<kB / 16, 256, 0, stream>>>(h, mask, trans, startv, endv, num, out);
}
